// SharedVocabCopyGenerator_25546465477119
// MI455X (gfx1250) — matrix operand order repaired and checked by reading, not compiled, not run on hardware
//
#include <hip/hip_runtime.h>
#include <hip/hip_bf16.h>
#include <float.h>

// ---------------------------------------------------------------------------
// SharedVocabCopyGenerator for MI455X (gfx1250, wave32, WMMA)
//   pass0: convert W (f32) -> bf16 copy in d_ws (memory-bound)
//   pass1: bf16 WMMA GEMM  logits = hidden @ W^T + b  -> d_out (scratch)
//          2 column tiles / iteration, double-buffered B fragments
//   pass2: combine partials across chunks, compute p_copy = sigmoid(h.Wc+bc)
//   pass3: prob = exp(l-M)/S * (1-p) + LDS-scatter(attn*p, src); log(clip)
// ---------------------------------------------------------------------------

typedef __attribute__((ext_vector_type(16))) __bf16 v16bf;
typedef __attribute__((ext_vector_type(8)))  float  v8f;

#define N_ROWS   2048
#define D_K      512
#define V_VOCAB  32000
#define S_SRC    400
#define B_BATCH  32

#define ROW_TILE 16
#define KSTEPS   (D_K / 32)          // 16 wmma k-steps
#define CHUNK1   6400                // pass1 V-chunk: 400 tiles, 50/wave (even)
#define NCHUNK1  (V_VOCAB / CHUNK1)  // 5
#define NTILES1  (CHUNK1 / 16)       // 400
#define CHUNKF   800                 // pass3 V-chunk (LDS scatter buffer)
#define NCHUNKF  (V_VOCAB / CHUNKF)  // 40

#define W_ELEMS  ((size_t)V_VOCAB * D_K)   // 16,384,000

__device__ __forceinline__ unsigned short bfbits(float f) {
  unsigned u = __builtin_bit_cast(unsigned, f);
  return (unsigned short)((u + 0x7FFFu + ((u >> 16) & 1u)) >> 16);  // RNE
}
__device__ __forceinline__ __bf16 f2bf(float f) {
  unsigned short s = bfbits(f);
  return __builtin_bit_cast(__bf16, s);
}
__device__ __forceinline__ unsigned pack2(float lo, float hi) {
  return (unsigned)bfbits(lo) | ((unsigned)bfbits(hi) << 16);
}

// ---------------------------------------------------------------------------
// pass 0: W f32 -> bf16 (row-major, same layout). 8 elements per thread.
// ---------------------------------------------------------------------------
__global__ __launch_bounds__(256) void svcg_pass0_cvtW(
    const float* __restrict__ W, __bf16* __restrict__ Wb)
{
  size_t i = ((size_t)blockIdx.x * 256 + threadIdx.x) * 8;
  if (i >= W_ELEMS) return;
  float4 a = *(const float4*)(W + i);
  float4 b = *(const float4*)(W + i + 4);
  uint4 p;
  p.x = pack2(a.x, a.y);
  p.y = pack2(a.z, a.w);
  p.z = pack2(b.x, b.y);
  p.w = pack2(b.z, b.w);
  *(uint4*)(Wb + i) = p;
}

// ---------------------------------------------------------------------------
// shared epilogue: bias add, logit store, online (max,sumexp) update
// ---------------------------------------------------------------------------
__device__ __forceinline__ void tile_epilogue(
    const v8f& c, float bv, float* __restrict__ logits,
    int mBase, int hh, int l16, int vTile,
    float rmax[8], float rsum[8])
{
  #pragma unroll
  for (int r = 0; r < 8; ++r) {            // C layout: VGPR r -> row r + 8*half
    float x = c[r] + bv;
    logits[(size_t)(mBase + 8 * hh + r) * V_VOCAB + vTile + l16] = x;
    float nm = fmaxf(rmax[r], x);
    rsum[r] = rsum[r] * __expf(rmax[r] - nm) + __expf(x - nm);
    rmax[r] = nm;
  }
}

__device__ __forceinline__ void chunk_reduce_and_write(
    float rmax[8], float rsum[8], int wave, int hh, int l16, int tid,
    int mBase, int chunkIdx, float pmax[8][ROW_TILE], float psum[8][ROW_TILE],
    float* __restrict__ partials)
{
  #pragma unroll
  for (int r = 0; r < 8; ++r) {
    for (int mask = 1; mask <= 8; mask <<= 1) {
      float om = __shfl_xor(rmax[r], mask, 32);
      float os = __shfl_xor(rsum[r], mask, 32);
      float nm = fmaxf(rmax[r], om);
      rsum[r] = rsum[r] * __expf(rmax[r] - nm) + os * __expf(om - nm);
      rmax[r] = nm;
    }
  }
  if (l16 == 0) {
    #pragma unroll
    for (int r = 0; r < 8; ++r) {
      pmax[wave][8 * hh + r] = rmax[r];
      psum[wave][8 * hh + r] = rsum[r];
    }
  }
  __syncthreads();
  if (tid < ROW_TILE) {
    float M = -FLT_MAX, S = 0.f;
    for (int w = 0; w < 8; ++w) {
      float om = pmax[w][tid], os = psum[w][tid];
      float nm = fmaxf(M, om);
      S = S * __expf(M - nm) + os * __expf(om - nm);
      M = nm;
    }
    size_t idx = ((size_t)chunkIdx * N_ROWS + (mBase + tid)) * 2;
    partials[idx] = M;
    partials[idx + 1] = S;
  }
}

// build A fragments from LDS-staged f32 hidden tile (cold path, once per WG)
__device__ __forceinline__ void build_a_frags(
    const float* hid /*[ROW_TILE*D_K] flat*/, int hh, int l16, v16bf a[KSTEPS])
{
  for (int kk = 0; kk < KSTEPS; ++kk) {
    #pragma unroll
    for (int i = 0; i < 8; ++i) {
      int k0 = kk * 32 + 16 * hh + 2 * i;
      float2 f = *(const float2*)&hid[l16 * D_K + k0];
      a[kk][2 * i]     = f2bf(f.x);
      a[kk][2 * i + 1] = f2bf(f.y);
    }
  }
}

// ---------------------------------------------------------------------------
// pass 1 (fast path): WMMA GEMM from pre-converted bf16 W
// grid = (N_ROWS/16, NCHUNK1), block = 256 (8 waves)
// ---------------------------------------------------------------------------
__global__ __launch_bounds__(256) void svcg_pass1_gemm_bf(
    const float* __restrict__ hidden, const __bf16* __restrict__ Wb,
    const float* __restrict__ bias, float* __restrict__ logits,
    float* __restrict__ partials)
{
  __shared__ float hid[ROW_TILE * D_K];    // 32 KB staged hidden tile
  __shared__ float pmax[8][ROW_TILE];
  __shared__ float psum[8][ROW_TILE];

  const int tid  = threadIdx.x;
  const int wave = tid >> 5;
  const int lane = tid & 31;
  const int hh   = lane >> 4;
  const int l16  = lane & 15;
  const int mBase      = blockIdx.x * ROW_TILE;
  const int chunkStart = blockIdx.y * CHUNK1;

  // hidden tile is a single contiguous 32KB block -> float4 copy
  {
    const float* srcp = hidden + (size_t)mBase * D_K;
    for (int i = tid * 4; i < ROW_TILE * D_K; i += 256 * 4)
      *(float4*)&hid[i] = *(const float4*)&srcp[i];
  }
  __syncthreads();

  v16bf a[KSTEPS];
  build_a_frags(hid, hh, l16, a);

  float rmax[8], rsum[8];
  #pragma unroll
  for (int r = 0; r < 8; ++r) { rmax[r] = -FLT_MAX; rsum[r] = 0.f; }

  // two column tiles (j, j+8) per iteration; double-buffered B fragments
  for (int j = wave; j < NTILES1; j += 16) {
    const int vT0 = chunkStart + j * 16;
    const int vT1 = vT0 + 128;                     // tile j+8
    const __bf16* wr0 = Wb + (size_t)(vT0 + l16) * D_K + 16 * hh;
    const __bf16* wr1 = wr0 + (size_t)128 * D_K;
    if (j + 16 < NTILES1) {
      __builtin_prefetch(wr0 + (size_t)256 * D_K, 0, 1);
      __builtin_prefetch(wr1 + (size_t)256 * D_K, 0, 1);
    }

    v8f c0 = {}, c1 = {};
    v16bf b0 = *(const v16bf*)(wr0);
    v16bf b1 = *(const v16bf*)(wr1);
    #pragma unroll
    for (int kk = 0; kk < KSTEPS; ++kk) {
      v16bf n0, n1;
      if (kk + 1 < KSTEPS) {
        n0 = *(const v16bf*)(wr0 + (kk + 1) * 32);
        n1 = *(const v16bf*)(wr1 + (kk + 1) * 32);
      }
      c0 = __builtin_amdgcn_wmma_f32_16x16x32_bf16(false, a[kk], false, b0,
                                                   (short)0, c0, false, false);
      c1 = __builtin_amdgcn_wmma_f32_16x16x32_bf16(false, a[kk], false, b1,
                                                   (short)0, c1, false, false);
      b0 = n0;
      b1 = n1;
    }
    tile_epilogue(c0, bias[vT0 + l16], logits, mBase, hh, l16, vT0, rmax, rsum);
    tile_epilogue(c1, bias[vT1 + l16], logits, mBase, hh, l16, vT1, rmax, rsum);
  }

  chunk_reduce_and_write(rmax, rsum, wave, hh, l16, tid, mBase, blockIdx.y,
                         pmax, psum, partials);
}

// ---------------------------------------------------------------------------
// pass 1 (fallback): converts W tiles on the fly (used if ws too small)
// ---------------------------------------------------------------------------
__global__ __launch_bounds__(256) void svcg_pass1_gemm_cvt(
    const float* __restrict__ hidden, const float* __restrict__ W,
    const float* __restrict__ bias, float* __restrict__ logits,
    float* __restrict__ partials)
{
  __shared__ float hid[ROW_TILE * D_K];
  __shared__ float pmax[8][ROW_TILE];
  __shared__ float psum[8][ROW_TILE];

  const int tid  = threadIdx.x;
  const int wave = tid >> 5;
  const int lane = tid & 31;
  const int hh   = lane >> 4;
  const int l16  = lane & 15;
  const int mBase      = blockIdx.x * ROW_TILE;
  const int chunkStart = blockIdx.y * CHUNK1;

  {
    const float* srcp = hidden + (size_t)mBase * D_K;
    for (int i = tid * 4; i < ROW_TILE * D_K; i += 256 * 4)
      *(float4*)&hid[i] = *(const float4*)&srcp[i];
  }
  __syncthreads();

  v16bf a[KSTEPS];
  build_a_frags(hid, hh, l16, a);

  float rmax[8], rsum[8];
  #pragma unroll
  for (int r = 0; r < 8; ++r) { rmax[r] = -FLT_MAX; rsum[r] = 0.f; }

  for (int j = wave; j < NTILES1; j += 8) {
    const int vTile = chunkStart + j * 16;
    const float* wrow = W + (size_t)(vTile + l16) * D_K + 16 * hh;
    if (j + 8 < NTILES1) __builtin_prefetch(wrow + 128 * D_K, 0, 1);

    v8f c = {};
    #pragma unroll
    for (int kk = 0; kk < KSTEPS; ++kk) {
      const float4* wp = (const float4*)(wrow + kk * 32);
      float4 w0 = wp[0], w1 = wp[1], w2 = wp[2], w3 = wp[3];
      v16bf bf;
      bf[0]  = f2bf(w0.x); bf[1]  = f2bf(w0.y); bf[2]  = f2bf(w0.z); bf[3]  = f2bf(w0.w);
      bf[4]  = f2bf(w1.x); bf[5]  = f2bf(w1.y); bf[6]  = f2bf(w1.z); bf[7]  = f2bf(w1.w);
      bf[8]  = f2bf(w2.x); bf[9]  = f2bf(w2.y); bf[10] = f2bf(w2.z); bf[11] = f2bf(w2.w);
      bf[12] = f2bf(w3.x); bf[13] = f2bf(w3.y); bf[14] = f2bf(w3.z); bf[15] = f2bf(w3.w);
      c = __builtin_amdgcn_wmma_f32_16x16x32_bf16(false, a[kk], false, bf,
                                                  (short)0, c, false, false);
    }
    tile_epilogue(c, bias[vTile + l16], logits, mBase, hh, l16, vTile, rmax, rsum);
  }

  chunk_reduce_and_write(rmax, rsum, wave, hh, l16, tid, mBase, blockIdx.y,
                         pmax, psum, partials);
}

// ---------------------------------------------------------------------------
// pass 2: per-row combine of chunk partials + p_copy = sigmoid(h.Wc + bc)
// ---------------------------------------------------------------------------
__global__ __launch_bounds__(256) void svcg_pass2_reduce(
    const float* __restrict__ hidden, const float* __restrict__ Wc,
    const float* __restrict__ bc, const float* __restrict__ partials,
    float* __restrict__ rowstats)
{
  const int row  = blockIdx.x * 8 + (threadIdx.x >> 5);
  const int lane = threadIdx.x & 31;
  if (row >= N_ROWS) return;

  float dot = 0.f;
  for (int k = lane; k < D_K; k += 32)
    dot += hidden[(size_t)row * D_K + k] * Wc[k];
  for (int mask = 16; mask >= 1; mask >>= 1)
    dot += __shfl_xor(dot, mask, 32);

  if (lane == 0) {
    float p = 1.f / (1.f + __expf(-(dot + bc[0])));
    float M = -FLT_MAX, S = 0.f;
    for (int c = 0; c < NCHUNK1; ++c) {
      size_t idx = ((size_t)c * N_ROWS + row) * 2;
      float om = partials[idx], os = partials[idx + 1];
      float nm = fmaxf(M, om);
      S = S * __expf(M - nm) + os * __expf(om - nm);
      M = nm;
    }
    rowstats[row * 3]     = M;
    rowstats[row * 3 + 1] = S;
    rowstats[row * 3 + 2] = p;
  }
}

// ---------------------------------------------------------------------------
// pass 3: finalize — softmax scale, LDS scatter-add of copy dist, log(clip)
// ---------------------------------------------------------------------------
__global__ __launch_bounds__(256) void svcg_pass3_finalize(
    const float* __restrict__ attn, const long long* __restrict__ src,
    const float* __restrict__ rowstats, float* __restrict__ outbuf)
{
  __shared__ float addbuf[ROW_TILE * CHUNKF];   // 51.2 KB scatter accumulator
  __shared__ float sM[ROW_TILE], sInvS[ROW_TILE], sP[ROW_TILE];

  const int tid        = threadIdx.x;
  const int mBase      = blockIdx.x * ROW_TILE;
  const int chunkStart = blockIdx.y * CHUNKF;

  for (int i = tid; i < ROW_TILE * CHUNKF; i += 256) addbuf[i] = 0.f;
  if (tid < ROW_TILE) {
    int n = mBase + tid;
    sM[tid]    = rowstats[n * 3];
    sInvS[tid] = 1.f / rowstats[n * 3 + 1];
    sP[tid]    = rowstats[n * 3 + 2];
  }
  __syncthreads();

  for (int idx = tid; idx < ROW_TILE * S_SRC; idx += 256) {
    int r = idx / S_SRC, s = idx - r * S_SRC;
    int n = mBase + r;
    int bb = n & (B_BATCH - 1);
    int v  = (int)src[(size_t)s * B_BATCH + bb];
    int vl = v - chunkStart;
    if ((unsigned)vl < (unsigned)CHUNKF)
      atomicAdd(&addbuf[r * CHUNKF + vl], attn[(size_t)n * S_SRC + s] * sP[r]);
  }
  __syncthreads();

  for (int r = 0; r < ROW_TILE; ++r) {
    int n = mBase + r;
    float M = sM[r], invS = sInvS[r], q = 1.f - sP[r];
    size_t base = (size_t)n * V_VOCAB + chunkStart;
    for (int v = tid; v < CHUNKF; v += 256) {
      float l = outbuf[base + v];
      float prob = __expf(l - M) * invS * q + addbuf[r * CHUNKF + v];
      outbuf[base + v] = __logf(fmaxf(prob, 1e-20f));
    }
  }
}

// ---------------------------------------------------------------------------
extern "C" void kernel_launch(void* const* d_in, const int* in_sizes, int n_in,
                              void* d_out, int out_size, void* d_ws, size_t ws_size,
                              hipStream_t stream) {
  const float*     hidden = (const float*)d_in[0];
  const float*     attn   = (const float*)d_in[1];
  const long long* src    = (const long long*)d_in[2];
  const float*     W      = (const float*)d_in[3];
  const float*     b      = (const float*)d_in[4];
  const float*     Wc     = (const float*)d_in[5];
  const float*     bc     = (const float*)d_in[6];
  float* out = (float*)d_out;

  // ws layout: [Wb bf16: W_ELEMS*2 B][partials: NCHUNK1*N*2 f32][rowstats: N*3 f32]
  const size_t wbBytes    = W_ELEMS * sizeof(unsigned short);
  const size_t partFloats = (size_t)NCHUNK1 * N_ROWS * 2;
  const size_t statFloats = (size_t)N_ROWS * 3;
  const size_t needBig    = wbBytes + (partFloats + statFloats) * sizeof(float);
  const bool   bigWs      = ws_size >= needBig;   // constant -> deterministic

  __bf16* Wb       = (__bf16*)d_ws;
  float*  partials = bigWs ? (float*)((char*)d_ws + wbBytes) : (float*)d_ws;
  float*  rowstats = partials + partFloats;

  if (bigWs) {
    const int cvtBlocks = (int)((W_ELEMS / 8 + 255) / 256);   // 8 elems/thread
    svcg_pass0_cvtW<<<dim3(cvtBlocks), 256, 0, stream>>>(W, Wb);
    svcg_pass1_gemm_bf<<<dim3(N_ROWS / ROW_TILE, NCHUNK1), 256, 0, stream>>>(
        hidden, Wb, b, out, partials);
  } else {
    svcg_pass1_gemm_cvt<<<dim3(N_ROWS / ROW_TILE, NCHUNK1), 256, 0, stream>>>(
        hidden, W, b, out, partials);
  }
  svcg_pass2_reduce<<<dim3(N_ROWS / 8), 256, 0, stream>>>(
      hidden, Wc, bc, partials, rowstats);
  svcg_pass3_finalize<<<dim3(N_ROWS / ROW_TILE, NCHUNKF), 256, 0, stream>>>(
      attn, src, rowstats, out);
}
